// DensitySetAbstraction_82188494177007
// MI455X (gfx1250) — matrix operand order repaired and checked by reading, not compiled, not run on hardware
//
#include <hip/hip_runtime.h>
#include <hip/hip_bf16.h>

// Problem constants (match reference)
#define B_   16
#define N_   4096
#define S_   1024
#define K_   32
#define CF   64               // C_FEAT
#define M_   (B_ * S_ * K_)   // 524288 GEMM columns
#define RAD2 0.04f
#define EPS_ 1e-5f
#define CP0  96               // layer-0 input channels 3+64=67 padded to 96

typedef __attribute__((ext_vector_type(16))) __bf16 v16bf;
typedef __attribute__((ext_vector_type(8)))  __bf16 v8bf;
typedef __attribute__((ext_vector_type(8)))  float  v8f;

// ---------------------------------------------------------------------------
// WMMA packing helpers (CDNA5 ISA 7.12.2, 16-bit layouts, wave32)
// B-matrix (32x16), contiguous k order used here (not ISA table order): lanes 0-15 hold K=0..15 (col n=lane), lanes 16-31 hold
// K=16..31 (col n=lane-16); element j within a lane = K offset 0..15.
__device__ __forceinline__ size_t bpack(int c, int m, int KB) {
    int kb = c >> 5, kl = c & 31;
    int lane = (m & 15) + ((kl >> 4) << 4);
    int j = kl & 15;
    return ((((size_t)(m >> 4) * KB + kb) * 32 + lane) << 4) | (size_t)j;
}

// A-matrix (16x32), same contiguous k order: lane<16: row=lane, elems 0-15=K0-15
//                   lane>=16: row=lane-16, elems 0-15=K16-31
__device__ __forceinline__ size_t apack(int o, int c, int KB) {
    int ob = o >> 4, ol = o & 15, kb = c >> 5, kl = c & 31;
    int lane = ol + ((kl >> 4) << 4), j = kl & 15;
    return ((((size_t)ob * KB + kb) * 32 + lane) << 4) | (size_t)j;
}

// ---------------------------------------------------------------------------
__global__ void zero_stats_kernel(float* __restrict__ st) {
    int i = blockIdx.x * blockDim.x + threadIdx.x;
    if (i < 1024) st[i] = 0.0f;
}

// ---------------------------------------------------------------------------
// 1) Farthest-point sampling: one block per batch; points live in registers.
#define FPT 256
#define PPT (N_ / FPT)

__global__ __launch_bounds__(FPT) void fps_kernel(const float* __restrict__ xyz,
                                                  int* __restrict__ fps_idx,
                                                  float* __restrict__ new_xyz) {
    const int b = blockIdx.x;
    const int t = threadIdx.x;
    const float* xb = xyz + (size_t)b * N_ * 3;

    float px[PPT], py[PPT], pz[PPT], dist[PPT];
#pragma unroll
    for (int i = 0; i < PPT; i++) {
        int n = t * PPT + i;
        px[i] = xb[n * 3 + 0];
        py[i] = xb[n * 3 + 1];
        pz[i] = xb[n * 3 + 2];
        dist[i] = 3.4e38f;
    }

    __shared__ float rv[FPT / 32];
    __shared__ int   ri[FPT / 32];
    __shared__ float s_lx, s_ly, s_lz;
    __shared__ int   s_last;

    int last = 0;
    if (t == 0) {
        fps_idx[b * S_ + 0] = 0;
        new_xyz[((size_t)b * S_ + 0) * 3 + 0] = xb[0];
        new_xyz[((size_t)b * S_ + 0) * 3 + 1] = xb[1];
        new_xyz[((size_t)b * S_ + 0) * 3 + 2] = xb[2];
    }

    for (int s = 1; s < S_; s++) {
        if (t == 0) {
            s_lx = xb[last * 3 + 0];
            s_ly = xb[last * 3 + 1];
            s_lz = xb[last * 3 + 2];
        }
        __syncthreads();
        float lx = s_lx, ly = s_ly, lz = s_lz;

        float bestv = -1.0f;
        int   besti = 0;
#pragma unroll
        for (int i = 0; i < PPT; i++) {
            float dx = px[i] - lx, dy = py[i] - ly, dz = pz[i] - lz;
            float d = dx * dx + dy * dy + dz * dz;
            dist[i] = fminf(dist[i], d);
            if (dist[i] > bestv) { bestv = dist[i]; besti = t * PPT + i; }
        }
#pragma unroll
        for (int off = 16; off > 0; off >>= 1) {
            float ov = __shfl_down(bestv, off, 32);
            int   oi = __shfl_down(besti, off, 32);
            if (ov > bestv || (ov == bestv && oi < besti)) { bestv = ov; besti = oi; }
        }
        if ((t & 31) == 0) { rv[t >> 5] = bestv; ri[t >> 5] = besti; }
        __syncthreads();
        if (t == 0) {
            bestv = rv[0]; besti = ri[0];
            for (int w = 1; w < FPT / 32; w++) {
                if (rv[w] > bestv || (rv[w] == bestv && ri[w] < besti)) {
                    bestv = rv[w]; besti = ri[w];
                }
            }
            s_last = besti;
            fps_idx[b * S_ + s] = besti;
            new_xyz[((size_t)b * S_ + s) * 3 + 0] = xb[besti * 3 + 0];
            new_xyz[((size_t)b * S_ + s) * 3 + 1] = xb[besti * 3 + 1];
            new_xyz[((size_t)b * S_ + s) * 3 + 2] = xb[besti * 3 + 2];
        }
        __syncthreads();
        last = s_last;
    }
}

// ---------------------------------------------------------------------------
// 2) Gather sampled densities + scalar mean/var stats
__global__ __launch_bounds__(256) void dens_kernel(const float* __restrict__ density,
                                                   const int* __restrict__ fps_idx,
                                                   float* __restrict__ dens,
                                                   float* __restrict__ dstats) {
    __shared__ float s1[256], s2[256];
    int tid = blockIdx.x * blockDim.x + threadIdx.x;   // 0 .. B*S-1
    int b = tid >> 10;
    float v = density[(size_t)b * N_ + fps_idx[tid]];
    dens[tid] = v;
    s1[threadIdx.x] = v;
    s2[threadIdx.x] = v * v;
    __syncthreads();
    for (int off = 128; off > 0; off >>= 1) {
        if (threadIdx.x < off) {
            s1[threadIdx.x] += s1[threadIdx.x + off];
            s2[threadIdx.x] += s2[threadIdx.x + off];
        }
        __syncthreads();
    }
    if (threadIdx.x == 0) {
        atomicAdd(&dstats[0], s1[0]);
        atomicAdd(&dstats[1], s2[0]);
    }
}

// ---------------------------------------------------------------------------
// 3) Ball query: first K in-ball indices ascending, pad with first.
#define BQT 256
__global__ __launch_bounds__(BQT) void ball_query_kernel(const float* __restrict__ xyz,
                                                         const float* __restrict__ new_xyz,
                                                         int* __restrict__ gidx) {
    __shared__ float sx[N_], sy[N_], sz[N_];
    const int b = blockIdx.x / (S_ / BQT);
    const int chunk = blockIdx.x % (S_ / BQT);
    const float* xb = xyz + (size_t)b * N_ * 3;
    for (int n = threadIdx.x; n < N_; n += BQT) {
        sx[n] = xb[n * 3 + 0];
        sy[n] = xb[n * 3 + 1];
        sz[n] = xb[n * 3 + 2];
    }
    __syncthreads();

    const int s = chunk * BQT + threadIdx.x;
    const size_t q = (size_t)b * S_ + s;
    const float qx = new_xyz[q * 3 + 0];
    const float qy = new_xyz[q * 3 + 1];
    const float qz = new_xyz[q * 3 + 2];
    int* gi = gidx + q * K_;
    int cnt = 0, first = 0;
    for (int n = 0; n < N_ && cnt < K_; n++) {
        float dx = sx[n] - qx, dy = sy[n] - qy, dz = sz[n] - qz;
        if (dx * dx + dy * dy + dz * dz <= RAD2) {
            if (cnt == 0) first = n;
            gi[cnt++] = n;
        }
    }
    for (int k = cnt; k < K_; k++) gi[k] = first;
}

// ---------------------------------------------------------------------------
// 4) Build layer-0 input X0 (bf16, B-packed, 96 x M). One thread produces one
//    contiguous 16-channel slot (32-byte v16bf store).
#define GSLOTS (CP0 / 16)   // 6
__global__ __launch_bounds__(256) void group_kernel(const float* __restrict__ xyz,
                                                    const float* __restrict__ points,
                                                    const float* __restrict__ new_xyz,
                                                    const int* __restrict__ gidx,
                                                    __bf16* __restrict__ X0) {
    int tid = blockIdx.x * blockDim.x + threadIdx.x;
    if (tid >= M_ * GSLOTS) return;
    int slot = tid % GSLOTS;
    int m = tid / GSLOTS;
    int kb = slot >> 1, h = slot & 1;
    int c0 = kb * 32 + h * 16;
    int b = m >> 15;                 // m / (S*K)
    int s = (m >> 5) & (S_ - 1);     // (m / K) % S
    int j = gidx[m];
    const float* pb = points + (size_t)b * CF * N_ + j;
    v16bf out;
#pragma unroll
    for (int e = 0; e < 16; e++) {
        int c = c0 + e;
        float v;
        if (c < 3) {
            v = xyz[((size_t)b * N_ + j) * 3 + c] - new_xyz[((size_t)b * S_ + s) * 3 + c];
        } else if (c < 3 + CF) {
            v = pb[(size_t)(c - 3) * N_];
        } else {
            v = 0.0f;
        }
        out[e] = (__bf16)v;
    }
    *(v16bf*)(X0 + (((size_t)(m >> 4) * (CP0 / 32) + kb) * 32 + (m & 15) + 16 * h) * 16) = out;
}

// ---------------------------------------------------------------------------
// 5) Pack weights f32 (O x Cin) -> bf16 A-layout (zero-padded to Cpad)
__global__ __launch_bounds__(256) void wprep_kernel(const float* __restrict__ W,
                                                    __bf16* __restrict__ Ap,
                                                    int O, int Cin, int Cpad) {
    int tid = blockIdx.x * blockDim.x + threadIdx.x;
    if (tid >= O * Cpad) return;
    int c = tid % Cpad, o = tid / Cpad;
    float v = (c < Cin) ? W[o * Cin + c] : 0.0f;
    Ap[apack(o, c, Cpad >> 5)] = (__bf16)v;
}

// ---------------------------------------------------------------------------
// 6) GEMM: one wave computes one 16-row block x NT=4 column tiles.
//    A tiles loaded once into registers; 4 independent accumulator streams
//    let loads overlap WMMAs. Epilogue: one 16-B packed store per lane/tile.
template <int OB, int KB>
__global__ __launch_bounds__(256) void gemm_kernel(const __bf16* __restrict__ Ap,
                                                   const __bf16* __restrict__ Xp,
                                                   const float* __restrict__ bias,
                                                   __bf16* __restrict__ Yp) {
    constexpr int NT = 4;
    const int wid = (blockIdx.x * blockDim.x + threadIdx.x) >> 5;
    const int lane = threadIdx.x & 31;
    constexpr int MB = M_ / 16;
    if (wid >= OB * (MB / NT)) return;
    const int ob = wid % OB;
    const int mg = wid / OB;              // group of NT consecutive column tiles

    v16bf a[KB];
#pragma unroll
    for (int kb = 0; kb < KB; kb++) {
        a[kb] = *(const v16bf*)(Ap + ((((size_t)ob * KB + kb) * 32 + lane) << 4));
    }

    v8f acc[NT];
#pragma unroll
    for (int t = 0; t < NT; t++) acc[t] = (v8f){};

#pragma unroll
    for (int t = 0; t < NT; t++) {
        const size_t mb = (size_t)mg * NT + t;
#pragma unroll
        for (int kb = 0; kb < KB; kb++) {
            v16bf x = *(const v16bf*)(Xp + (((mb * KB + kb) * 32 + lane) << 4));
            acc[t] = __builtin_amdgcn_wmma_f32_16x16x32_bf16(false, a[kb], false, x,
                                                             (short)0, acc[t], false, false);
        }
    }

    const int n = lane & 15;
    const int hi = lane >> 4;
    const int o0 = ob * 16 + 8 * hi;      // first of 8 contiguous output channels
    constexpr int O = OB * 16;
    float bv[8];
#pragma unroll
    for (int i = 0; i < 8; i++) bv[i] = bias[o0 + i];

#pragma unroll
    for (int t = 0; t < NT; t++) {
        const int m = (mg * NT + t) * 16 + n;
        v8bf out;
#pragma unroll
        for (int i = 0; i < 8; i++) out[i] = (__bf16)(acc[t][i] + bv[i]);
        *(v8bf*)(Yp + bpack(o0, m, O >> 5)) = out;   // 16-byte aligned
    }
}

// ---------------------------------------------------------------------------
// 7) Per-channel sum / sum-of-squares. Channel = f(kb, lane-half, j) only, so
//    each lane owns one channel, accumulates in registers, 2 atomics at end.
//    Wave w: kb = w % KB, m-partition = w / KB (requires KB in {2,4}).
__global__ __launch_bounds__(256) void stats_kernel(const __bf16* __restrict__ Yp,
                                                    float* __restrict__ st, int O) {
    const int KB = O >> 5;
    const int WP = 8 / KB;
    const int t = threadIdx.x;
    const int w = t >> 5;
    const int lane = t & 31;
    const int j = lane & 15;
    const int h = lane >> 4;
    const int kb = w % KB;
    const int mpart = w / KB;
    const int c = kb * 32 + h * 16 + j;
    const int MBtot = M_ / 16;
    const int parts = gridDim.x * WP;
    float sum = 0.0f, sq = 0.0f;
    for (int mb = blockIdx.x * WP + mpart; mb < MBtot; mb += parts) {
        const __bf16* p = Yp + ((size_t)mb * KB + kb) * 512 + h * 256 + j;
#pragma unroll
        for (int ml = 0; ml < 16; ml++) {
            float v = (float)p[ml * 16];
            sum += v;
            sq += v * v;
        }
    }
    atomicAdd(&st[c], sum);
    atomicAdd(&st[O + c], sq);
}

// ---------------------------------------------------------------------------
// 8) In-place BatchNorm (training) + ReLU, 16 elements (32 B) per thread.
__global__ __launch_bounds__(256) void bn_apply_kernel(__bf16* __restrict__ Yp,
                                                       const float* __restrict__ st,
                                                       const float* __restrict__ g,
                                                       const float* __restrict__ bt,
                                                       int O) {
    const int KB = O >> 5;
    size_t i = (size_t)blockIdx.x * blockDim.x + threadIdx.x;   // vector index
    if (i >= ((size_t)O * M_) >> 4) return;
    int lane = (int)(i & 31);
    int kb = (int)((i >> 5) % (size_t)KB);
    int c0 = kb * 32 + ((lane >> 4) << 4);
    v16bf v = *(const v16bf*)(Yp + (i << 4));
    v16bf r;
#pragma unroll
    for (int j = 0; j < 16; j++) {
        int c = c0 + j;
        float mean = st[c] * (1.0f / M_);
        float var = st[O + c] * (1.0f / M_) - mean * mean;
        float x = (float)v[j];
        x = (x - mean) * rsqrtf(var + EPS_) * g[c] + bt[c];
        r[j] = (__bf16)fmaxf(x, 0.0f);
    }
    *(v16bf*)(Yp + (i << 4)) = r;
}

// ---------------------------------------------------------------------------
// 9) Final: max over K, multiply by analytic density-branch weight.
//    o-fastest mapping -> the 32 reduction loads are wave-contiguous.
__global__ __launch_bounds__(256) void final_kernel(const __bf16* __restrict__ X3,
                                                    const float* __restrict__ dens,
                                                    const float* __restrict__ dstats,
                                                    const float* __restrict__ Wd,
                                                    const float* __restrict__ bd,
                                                    const float* __restrict__ gd,
                                                    const float* __restrict__ btd,
                                                    float* __restrict__ out_np) {
    const int O = 128;
    int tid = blockIdx.x * blockDim.x + threadIdx.x;
    if (tid >= B_ * O * S_) return;
    int o = tid & (O - 1);
    int s = (tid >> 7) & (S_ - 1);
    int b = tid >> 17;
    int kb = o >> 5, h = (o >> 4) & 1, j = o & 15;
    int mb0 = (((b << 10) + s) << 5) >> 4;           // (m0)/16, m0 = (b*S+s)*K
    const __bf16* base = X3 + (size_t)kb * 512 + h * 256 + j;
    float mx = -3.4e38f;
#pragma unroll
    for (int k = 0; k < K_; k++) {
        size_t f = ((size_t)(mb0 + (k >> 4)) * 4) * 512 + (size_t)(k & 15) * 16;
        mx = fmaxf(mx, (float)base[f]);
    }
    float mu = dstats[0] * (1.0f / (B_ * S_));
    float var = dstats[1] * (1.0f / (B_ * S_)) - mu * mu;
    float w = Wd[o];                                  // Wd shape (128,1)
    float y = w * dens[(b << 10) + s] + bd[o];
    float ymu = w * mu + bd[o];
    float yvar = w * w * var;
    float dw = fmaxf((y - ymu) * rsqrtf(yvar + EPS_) * gd[o] + btd[o], 0.0f);
    out_np[(((size_t)b * O + o) << 10) + s] = mx * dw;
}

// ---------------------------------------------------------------------------
extern "C" void kernel_launch(void* const* d_in, const int* in_sizes, int n_in,
                              void* d_out, int out_size, void* d_ws, size_t ws_size,
                              hipStream_t stream) {
    const float* xyz     = (const float*)d_in[0];
    const float* points  = (const float*)d_in[1];
    const float* density = (const float*)d_in[2];
    const float* W0 = (const float*)d_in[3],  *b0 = (const float*)d_in[4];
    const float* g0 = (const float*)d_in[5],  *bt0 = (const float*)d_in[6];
    const float* W1 = (const float*)d_in[7],  *b1 = (const float*)d_in[8];
    const float* g1 = (const float*)d_in[9],  *bt1 = (const float*)d_in[10];
    const float* W2 = (const float*)d_in[11], *b2 = (const float*)d_in[12];
    const float* g2 = (const float*)d_in[13], *bt2 = (const float*)d_in[14];
    const float* Wd = (const float*)d_in[15], *bd = (const float*)d_in[16];
    const float* gd = (const float*)d_in[17], *btd = (const float*)d_in[18];

    float* out_xyz = (float*)d_out;                       // (B,S,3)
    float* out_np  = (float*)d_out + (size_t)B_ * S_ * 3; // (B,128,S)

    // workspace layout (256B-aligned regions)
    char* ws = (char*)d_ws;
    size_t off = 0;
    float* stats = (float*)(ws + off); off += 4096;       // 1024 floats
    float* dstats = stats;            // [0..2)
    float* st0 = stats + 128;         // 2*64
    float* st1 = stats + 256;         // 2*64
    float* st2 = stats + 384;         // 2*128
    int* fps_idx = (int*)(ws + off);  off += (size_t)B_ * S_ * 4;
    int* gidx    = (int*)(ws + off);  off += (size_t)B_ * S_ * K_ * 4;
    float* dens  = (float*)(ws + off); off += (size_t)B_ * S_ * 4;
    __bf16* Wp0 = (__bf16*)(ws + off); off += (size_t)64 * CP0 * 2;
    __bf16* Wp1 = (__bf16*)(ws + off); off += (size_t)64 * 64 * 2;
    __bf16* Wp2 = (__bf16*)(ws + off); off += (size_t)128 * 64 * 2;
    __bf16* bufA = (__bf16*)(ws + off); off += (size_t)CP0 * M_ * 2;   // X0 / Y1,X2
    __bf16* bufB = (__bf16*)(ws + off); off += (size_t)128 * M_ * 2;   // Y0,X1 / Y2,X3
    if (ws_size < off) return;   // workspace too small; avoid OOB

    zero_stats_kernel<<<4, 256, 0, stream>>>(stats);
    fps_kernel<<<B_, FPT, 0, stream>>>(xyz, fps_idx, out_xyz);
    dens_kernel<<<(B_ * S_) / 256, 256, 0, stream>>>(density, fps_idx, dens, dstats);
    ball_query_kernel<<<B_ * (S_ / BQT), BQT, 0, stream>>>(xyz, out_xyz, gidx);
    group_kernel<<<(M_ * GSLOTS) / 256, 256, 0, stream>>>(xyz, points, out_xyz, gidx, bufA);
    wprep_kernel<<<(64 * CP0 + 255) / 256, 256, 0, stream>>>(W0, Wp0, 64, 67, CP0);
    wprep_kernel<<<(64 * 64 + 255) / 256, 256, 0, stream>>>(W1, Wp1, 64, 64, 64);
    wprep_kernel<<<(128 * 64 + 255) / 256, 256, 0, stream>>>(W2, Wp2, 128, 64, 64);

    const int MB = M_ / 16;
    // Layer 0: (64 x 96) x (96 x M) -> bufB
    gemm_kernel<4, 3><<<(4 * (MB / 4)) * 32 / 256, 256, 0, stream>>>(Wp0, bufA, b0, bufB);
    stats_kernel<<<1024, 256, 0, stream>>>(bufB, st0, 64);
    bn_apply_kernel<<<(unsigned)(((size_t)64 * M_ / 16) / 256), 256, 0, stream>>>(bufB, st0, g0, bt0, 64);
    // Layer 1: (64 x 64) x (64 x M) -> bufA
    gemm_kernel<4, 2><<<(4 * (MB / 4)) * 32 / 256, 256, 0, stream>>>(Wp1, bufB, b1, bufA);
    stats_kernel<<<1024, 256, 0, stream>>>(bufA, st1, 64);
    bn_apply_kernel<<<(unsigned)(((size_t)64 * M_ / 16) / 256), 256, 0, stream>>>(bufA, st1, g1, bt1, 64);
    // Layer 2: (128 x 64) x (64 x M) -> bufB
    gemm_kernel<8, 2><<<(8 * (MB / 4)) * 32 / 256, 256, 0, stream>>>(Wp2, bufA, b2, bufB);
    stats_kernel<<<2048, 256, 0, stream>>>(bufB, st2, 128);
    bn_apply_kernel<<<(unsigned)(((size_t)128 * M_ / 16) / 256), 256, 0, stream>>>(bufB, st2, g2, bt2, 128);

    final_kernel<<<(B_ * 128 * S_) / 256, 256, 0, stream>>>(bufB, dens, dstats, Wd, bd, gd, btd, out_np);
}
